// Mamba3Layer_16630113370812
// MI455X (gfx1250) — matrix operand order repaired and checked by reading, not compiled, not run on hardware
//
#include <hip/hip_runtime.h>
#include <hip/hip_bf16.h>
#include <cstdint>

// ---------------------------------------------------------------------------
// Model constants
// ---------------------------------------------------------------------------
#define D_MODEL   1024
#define D_STATE   64
#define D_INNER   2048
#define N_HEADS   32
#define HEAD_DIM  64
#define D_PROJ    4320          // 2048 + 2048 + 64 + 64 + 32 + 32 + 32
#define BATCH     2
#define SEQ_T     4096
#define BT        (BATCH * SEQ_T)   // 8192 tokens
#define OFF_Z     0
#define OFF_X     2048
#define OFF_B     4096
#define OFF_C     4160
#define OFF_DT    4224
#define OFF_LAM   4256
#define OFF_TH    4288
#define EPS_RMS   1.1920929e-07f
#define SCALE_Y   0.125f        // 64^-0.5

typedef __attribute__((ext_vector_type(16))) __bf16 v16bf;
typedef __attribute__((ext_vector_type(8)))  float  v8f;

// ---------------------------------------------------------------------------
// Async global->LDS support (gfx1250). Guarded so a missing builtin falls
// back to the synchronous load/ds-store path. The builtin's parameters are
// AS(1)/AS(3) pointers to a 16-byte int vector (per the toolchain diagnostic).
// ---------------------------------------------------------------------------
#if __has_builtin(__builtin_amdgcn_global_load_async_to_lds_b128) && \
    __has_builtin(__builtin_amdgcn_s_wait_asynccnt)
#define USE_ASYNC_LDS 1
#else
#define USE_ASYNC_LDS 0
#endif

#if USE_ASYNC_LDS
typedef int v4i __attribute__((__vector_size__(16)));
typedef __attribute__((address_space(1))) v4i* gp_v4i;   // global (AS1) int4*
typedef __attribute__((address_space(3))) v4i* lp_v4i;   // LDS    (AS3) int4*

__device__ __forceinline__ void async_ld_b128(void* lds, const void* g) {
    __builtin_amdgcn_global_load_async_to_lds_b128(
        (gp_v4i)(uintptr_t)g,   // global pointers: same representation as generic
        (lp_v4i)lds,            // addrspacecast generic -> LDS
        0, 0);
}
__device__ __forceinline__ void wait_async0() {
    __builtin_amdgcn_s_wait_asynccnt(0);
}
#endif

__device__ __forceinline__ uint16_t f2bf(float f) {
    union { float f; uint32_t u; } c; c.f = f;
    uint32_t r = (c.u + 0x7FFFu + ((c.u >> 16) & 1u)) >> 16;   // RNE
    return (uint16_t)r;
}

// ---------------------------------------------------------------------------
// K0: f32 -> bf16 conversion (x, W_in, W_out)
// ---------------------------------------------------------------------------
__global__ void f32_to_bf16_kernel(const float* __restrict__ in,
                                   uint16_t* __restrict__ out, size_t n) {
    size_t i  = (size_t)blockIdx.x * blockDim.x + threadIdx.x;
    size_t st = (size_t)gridDim.x * blockDim.x;
    for (; i < n; i += st) out[i] = f2bf(in[i]);
}

// ---------------------------------------------------------------------------
// K1/K5: bf16 WMMA GEMM:  C[M][N] (f32) = A[M][K] * B[N][K]^T
// Block tile 128x128, BK=64, 256 threads = 8 waves (2x4), wave tile 64x32,
// 4x2 of v_wmma_f32_16x16x32_bf16 per wave per k-step. Double-buffered LDS
// staged with GLOBAL_LOAD_ASYNC_TO_LDS_B128 (ASYNCcnt) when available.
// ---------------------------------------------------------------------------
#define BM  128
#define BN  128
#define BK  64
#define LDT 72                  // halves stride (+8 pad -> 144B rows, 16B aligned)

__global__ __launch_bounds__(256)
void gemm_bf16_kernel(const uint16_t* __restrict__ A,   // [M][K] row-major bf16
                      const uint16_t* __restrict__ Bw,  // [N][K] row-major bf16
                      float* __restrict__ C,            // [M][N] f32
                      int M, int N, int K) {
#if USE_ASYNC_LDS
    __shared__ uint16_t As[2][BM * LDT];
    __shared__ uint16_t Bs[2][BN * LDT];
#else
    __shared__ uint16_t As[1][BM * LDT];
    __shared__ uint16_t Bs[1][BN * LDT];
#endif

    const int tid  = threadIdx.x;
    const int lane = tid & 31;
    const int wave = tid >> 5;
    const int wm   = wave >> 2;        // 0..1  (M direction)
    const int wn   = wave & 3;         // 0..3  (N direction)
    const int g    = lane >> 4;        // lane group (K split)
    const int l16  = lane & 15;

    const int bm = blockIdx.y * BM;
    const int bn = blockIdx.x * BN;

    v8f acc[4][2];
    const v8f vzero = {0.f,0.f,0.f,0.f,0.f,0.f,0.f,0.f};
#pragma unroll
    for (int i = 0; i < 4; ++i)
#pragma unroll
        for (int j = 0; j < 2; ++j) acc[i][j] = vzero;

#if USE_ASYNC_LDS
    // ---- async tile stage: 4x b128 for A, 4x b128 for B per thread ----
    auto stage = [&](int k0, int buf) {
#pragma unroll
        for (int i = 0; i < 4; ++i) {
            int idx = tid + i * 256;           // 1024 chunks of 8 halves
            int row = idx >> 3;
            int col = (idx & 7) * 8;
            const uint16_t* srcA = A + (size_t)(bm + row) * K + k0 + col;
            async_ld_b128(&As[buf][row * LDT + col], srcA);
            int n = bn + row;
            if (n >= N) n = N - 1;             // clamp: keeps EXEC full, junk cols dropped at store
            const uint16_t* srcB = Bw + (size_t)n * K + k0 + col;
            async_ld_b128(&Bs[buf][row * LDT + col], srcB);
            if (i == 0 && k0 + BK < K) __builtin_prefetch(srcA + BK, 0, 2);
        }
    };

    stage(0, 0);
    int buf = 0;
    for (int k0 = 0; k0 < K; k0 += BK) {
        wait_async0();                         // my stage(k0) loads landed in LDS
        __syncthreads();                       // everyone's landed; prev compute done
        if (k0 + BK < K) stage(k0 + BK, buf ^ 1);
        const uint16_t* Asb = As[buf];
        const uint16_t* Bsb = Bs[buf];
#else
    for (int k0 = 0; k0 < K; k0 += BK) {
        // ---- synchronous fallback staging ----
#pragma unroll
        for (int i = 0; i < 4; ++i) {
            int idx = tid + i * 256;
            int row = idx >> 3;
            int col = (idx & 7) * 8;
            const uint16_t* src = A + (size_t)(bm + row) * K + k0 + col;
            if (k0 + BK < K) __builtin_prefetch(src + BK, 0, 2);
            *(uint4*)(&As[0][row * LDT + col]) = *(const uint4*)src;
        }
#pragma unroll
        for (int i = 0; i < 4; ++i) {
            int idx = tid + i * 256;
            int row = idx >> 3;
            int col = (idx & 7) * 8;
            int n   = bn + row;
            uint4 v = make_uint4(0u, 0u, 0u, 0u);
            if (n < N) v = *(const uint4*)(Bw + (size_t)n * K + k0 + col);
            *(uint4*)(&Bs[0][row * LDT + col]) = v;
        }
        __syncthreads();
        const uint16_t* Asb = As[0];
        const uint16_t* Bsb = Bs[0];
#endif

#pragma unroll
        for (int kk = 0; kk < BK; kk += 32) {
            // B fragments (32x16, per-lane K ranges g*8..g*8+7 and 16+g*8..16+g*8+7)
            v16bf bfrag[2];
#pragma unroll
            for (int nt = 0; nt < 2; ++nt) {
                const uint16_t* p = &Bsb[(wn * 32 + nt * 16 + l16) * LDT + kk + g * 8];
                union { v16bf v; uint4 q[2]; } u;
                u.q[0] = *(const uint4*)(p);
                u.q[1] = *(const uint4*)(p + 16);
                bfrag[nt] = u.v;
            }
#pragma unroll
            for (int mt = 0; mt < 4; ++mt) {
                // A fragment (16x32): pair v holds K = (v>=4?16:0) + g*8 + (v&3)*2
                const uint16_t* pa = &Asb[(wm * 64 + mt * 16 + l16) * LDT + kk];
                union { v16bf v; uint32_t w[8]; } ua;
#pragma unroll
                for (int v2 = 0; v2 < 8; ++v2) {
                    int kp = ((v2 >= 4) ? 16 : 0) + g * 8 + (v2 & 3) * 2;
                    ua.w[v2] = *(const uint32_t*)(pa + kp);
                }
#pragma unroll
                for (int nt = 0; nt < 2; ++nt) {
                    acc[mt][nt] = __builtin_amdgcn_wmma_f32_16x16x32_bf16(
                        false, ua.v, false, bfrag[nt],
                        (short)0, acc[mt][nt], false, false);
                }
            }
        }
#if USE_ASYNC_LDS
        buf ^= 1;
#else
        __syncthreads();
#endif
    }

    // ---- store: C/D layout — VGPR r: row = g*8 + r, col = l16 ----
#pragma unroll
    for (int mt = 0; mt < 4; ++mt)
#pragma unroll
        for (int nt = 0; nt < 2; ++nt) {
            int col = bn + wn * 32 + nt * 16 + l16;
            if (col < N) {
#pragma unroll
                for (int r = 0; r < 8; ++r) {
                    int row = bm + wm * 64 + mt * 16 + g * 8 + r;
                    C[(size_t)row * N + col] = acc[mt][nt][r];
                }
            }
        }
}

// ---------------------------------------------------------------------------
// K2a: per-token prep: dt/alpha/lambda, dt_mean, RMS-norm of B and C
// ---------------------------------------------------------------------------
__global__ void prep_pos_kernel(const float* __restrict__ proj,
                                const float* __restrict__ A_log,
                                const float* __restrict__ dt_bias,
                                const float* __restrict__ Bbias,
                                const float* __restrict__ Cbias,
                                float* __restrict__ Bn, float* __restrict__ Cn,
                                float* __restrict__ alpha, float* __restrict__ dtv,
                                float* __restrict__ lam, float* __restrict__ dtmean) {
    const int pos = blockIdx.x;
    const int tid = threadIdx.x;            // 64
    const float* row = proj + (size_t)pos * D_PROJ;
    __shared__ float red[64];

    float d = 0.f;
    if (tid < 32) {
        float draw = row[OFF_DT + tid] + dt_bias[tid];
        float sp = (draw > 20.f) ? draw : log1pf(expf(draw));
        d = fminf(fmaxf(sp, 1e-4f), 0.5f);
        dtv[(size_t)pos * 32 + tid]   = d;
        alpha[(size_t)pos * 32 + tid] = expf(-expf(A_log[tid]) * d);
        float lr = row[OFF_LAM + tid];
        lam[(size_t)pos * 32 + tid]   = 1.f / (1.f + expf(-lr));
    }
    red[tid] = (tid < 32) ? d : 0.f;
    __syncthreads();
    for (int s = 32; s > 0; s >>= 1) { if (tid < s) red[tid] += red[tid + s]; __syncthreads(); }
    if (tid == 0) dtmean[pos] = red[0] * (1.f / 32.f);
    __syncthreads();

    float b = row[OFF_B + tid];
    red[tid] = b * b; __syncthreads();
    for (int s = 32; s > 0; s >>= 1) { if (tid < s) red[tid] += red[tid + s]; __syncthreads(); }
    float msB = red[0] * (1.f / 64.f);
    __syncthreads();

    float c = row[OFF_C + tid];
    red[tid] = c * c; __syncthreads();
    for (int s = 32; s > 0; s >>= 1) { if (tid < s) red[tid] += red[tid + s]; __syncthreads(); }
    float msC = red[0] * (1.f / 64.f);

    Bn[(size_t)pos * 64 + tid] = b * rsqrtf(msB + EPS_RMS) * Bbias[tid];
    Cn[(size_t)pos * 64 + tid] = c * rsqrtf(msC + EPS_RMS) * Cbias[tid];
}

// ---------------------------------------------------------------------------
// K2b: serial cumulative angle per (batch, channel): 2 blocks x 32 threads
// ---------------------------------------------------------------------------
__global__ void cumsum_theta_kernel(const float* __restrict__ proj,
                                    const float* __restrict__ dtmean,
                                    float* __restrict__ cosA, float* __restrict__ sinA) {
    const int b = blockIdx.x;
    const int c = threadIdx.x;              // 32
    float ang = 0.f;
    for (int t = 0; t < SEQ_T; ++t) {
        size_t pos = (size_t)b * SEQ_T + t;
        ang -= dtmean[pos] * proj[pos * D_PROJ + OFF_TH + c];
        cosA[pos * 32 + c] = cosf(ang);
        sinA[pos * 32 + c] = sinf(ang);
    }
}

// ---------------------------------------------------------------------------
// K2c: RoPE pairs applied in place to Bn, Cn
// ---------------------------------------------------------------------------
__global__ void rope_kernel(const float* __restrict__ cosA, const float* __restrict__ sinA,
                            float* __restrict__ Bn, float* __restrict__ Cn) {
    size_t i  = (size_t)blockIdx.x * blockDim.x + threadIdx.x;
    size_t st = (size_t)gridDim.x * blockDim.x;
    size_t n  = (size_t)BT * 32;
    for (; i < n; i += st) {
        size_t pos = i >> 5; int c = (int)(i & 31);
        float cs = cosA[i], sn = sinA[i];
        float b1 = Bn[pos * 64 + c], b2 = Bn[pos * 64 + c + 32];
        Bn[pos * 64 + c]      = b1 * cs - b2 * sn;
        Bn[pos * 64 + c + 32] = b1 * sn + b2 * cs;
        float c1 = Cn[pos * 64 + c], c2 = Cn[pos * 64 + c + 32];
        Cn[pos * 64 + c]      = c1 * cs - c2 * sn;
        Cn[pos * 64 + c + 32] = c1 * sn + c2 * cs;
    }
}

// ---------------------------------------------------------------------------
// K2d: SiLU in place on x_raw region of proj
// ---------------------------------------------------------------------------
__global__ void silu_x_kernel(float* __restrict__ proj) {
    size_t i  = (size_t)blockIdx.x * blockDim.x + threadIdx.x;
    size_t st = (size_t)gridDim.x * blockDim.x;
    size_t n  = (size_t)BT * D_INNER;
    for (; i < n; i += st) {
        size_t pos = i >> 11; int c = (int)(i & 2047);
        float* p = proj + pos * D_PROJ + OFF_X + c;
        float v = *p;
        *p = v / (1.f + expf(-v));
    }
}

// ---------------------------------------------------------------------------
// K3: sequential SSM scan. One block per (batch, head). 256 threads hold the
// 64x64 f32 state in registers (16 each: fixed d = tid&63, s = (tid>>6)*16+j).
// Chunk staging goes straight to LDS via async-to-LDS b128 when available.
// ---------------------------------------------------------------------------
#define CH 32
__global__ __launch_bounds__(256)
void scan_kernel(const float* __restrict__ proj,     // silu'd x at OFF_X
                 const float* __restrict__ Brot, const float* __restrict__ Crot,
                 const float* __restrict__ alpha, const float* __restrict__ dtv,
                 const float* __restrict__ lam,
                 float* __restrict__ y) {
    const int bh = blockIdx.x;
    const int b  = bh >> 5;
    const int h  = bh & 31;
    const int tid = threadIdx.x;
    const int d   = tid & 63;
    const int sg  = tid >> 6;           // 0..3
    const int s0  = sg * 16;

    __shared__ float xl[CH + 1][HEAD_DIM];   // rows: t0-1 .. t0+CH-1
    __shared__ float Bl[CH + 1][D_STATE];
    __shared__ float Cl[CH][D_STATE];
    __shared__ float sa[CH], sd[CH], sl[CH];
    __shared__ float part[4][CH][HEAD_DIM];  // 32 KB partial y

    float hst[16];
#pragma unroll
    for (int j = 0; j < 16; ++j) hst[j] = 0.f;

    for (int t0 = 0; t0 < SEQ_T; t0 += CH) {
#if USE_ASYNC_LDS
        for (int i = tid; i < (CH + 1) * 16; i += 256) {   // 4-float chunks
            int r = i >> 4, d4 = (i & 15) * 4;
            int t = t0 - 1 + r;
            if (t >= 0) {
                size_t pos = (size_t)b * SEQ_T + t;
                async_ld_b128(&xl[r][d4], &proj[pos * D_PROJ + OFF_X + h * HEAD_DIM + d4]);
                async_ld_b128(&Bl[r][d4], &Brot[pos * 64 + d4]);
            } else {
                *(float4*)&xl[r][d4] = make_float4(0.f, 0.f, 0.f, 0.f);
                *(float4*)&Bl[r][d4] = make_float4(0.f, 0.f, 0.f, 0.f);
            }
        }
        for (int i = tid; i < CH * 16; i += 256) {
            int r = i >> 4, d4 = (i & 15) * 4;
            size_t pos = (size_t)b * SEQ_T + t0 + r;
            async_ld_b128(&Cl[r][d4], &Crot[pos * 64 + d4]);
        }
#else
        for (int i = tid; i < (CH + 1) * 64; i += 256) {
            int r = i >> 6, dd = i & 63;
            int t = t0 - 1 + r;
            float xv = 0.f, bv = 0.f;
            if (t >= 0) {
                size_t pos = (size_t)b * SEQ_T + t;
                xv = proj[pos * D_PROJ + OFF_X + h * HEAD_DIM + dd];
                bv = Brot[pos * 64 + dd];
            }
            xl[r][dd] = xv;
            Bl[r][dd] = bv;
        }
        for (int i = tid; i < CH * 64; i += 256) {
            int r = i >> 6, dd = i & 63;
            size_t pos = (size_t)b * SEQ_T + t0 + r;
            Cl[r][dd] = Crot[pos * 64 + dd];
        }
#endif
        if (tid < CH) {
            size_t pos = (size_t)b * SEQ_T + t0 + tid;
            sa[tid] = alpha[pos * 32 + h];
            sd[tid] = dtv[pos * 32 + h];
            sl[tid] = lam[pos * 32 + h];
        }
#if USE_ASYNC_LDS
        wait_async0();
#endif
        __syncthreads();

        for (int tt = 0; tt < CH; ++tt) {
            float a = sa[tt], dt = sd[tt], l = sl[tt];
            float cg = dt * l * xl[tt + 1][d];              // dt * lam * x_t
            float cb = dt * (1.f - l) * a * xl[tt][d];      // dt * (1-lam) * alpha * x_{t-1}
            float yp = 0.f;
#pragma unroll
            for (int j = 0; j < 16; ++j) {
                float hh = fmaf(a, hst[j], fmaf(cg, Bl[tt + 1][s0 + j], cb * Bl[tt][s0 + j]));
                hst[j] = hh;
                yp = fmaf(hh, Cl[tt][s0 + j], yp);
            }
            part[sg][tt][d] = yp;
        }
        __syncthreads();

        for (int i = tid; i < CH * 64; i += 256) {
            int r = i >> 6, dd = i & 63;
            float v = (part[0][r][dd] + part[1][r][dd] +
                       part[2][r][dd] + part[3][r][dd]) * SCALE_Y;
            size_t pos = (size_t)b * SEQ_T + t0 + r;
            y[pos * D_INNER + h * HEAD_DIM + dd] = v;
        }
        __syncthreads();
    }
}

// ---------------------------------------------------------------------------
// K4: y = rms_norm(y * silu(z)) -> bf16.  Block per token, 256 threads.
// ---------------------------------------------------------------------------
__global__ __launch_bounds__(256)
void gate_norm_kernel(const float* __restrict__ y, const float* __restrict__ proj,
                      uint16_t* __restrict__ yb) {
    const int pos = blockIdx.x;
    const int tid = threadIdx.x;
    __shared__ float red[256];
    const float* yr = y + (size_t)pos * D_INNER;
    const float* zr = proj + (size_t)pos * D_PROJ + OFF_Z;

    float vals[8];
    float ss = 0.f;
#pragma unroll
    for (int i = 0; i < 8; ++i) {
        int idx = tid + i * 256;
        float z = zr[idx];
        float v = yr[idx] * (z / (1.f + expf(-z)));
        vals[i] = v;
        ss += v * v;
    }
    red[tid] = ss; __syncthreads();
    for (int s = 128; s > 0; s >>= 1) { if (tid < s) red[tid] += red[tid + s]; __syncthreads(); }
    float inv = rsqrtf(red[0] * (1.f / (float)D_INNER) + EPS_RMS);
#pragma unroll
    for (int i = 0; i < 8; ++i)
        yb[(size_t)pos * D_INNER + tid + i * 256] = f2bf(vals[i] * inv);
}

// ---------------------------------------------------------------------------
// Host launch
// ---------------------------------------------------------------------------
extern "C" void kernel_launch(void* const* d_in, const int* in_sizes, int n_in,
                              void* d_out, int out_size, void* d_ws, size_t ws_size,
                              hipStream_t stream) {
    const float* x       = (const float*)d_in[0];
    const float* W_in    = (const float*)d_in[1];
    const float* A_log   = (const float*)d_in[2];
    const float* dt_bias = (const float*)d_in[3];
    const float* Bbias   = (const float*)d_in[4];
    const float* Cbias   = (const float*)d_in[5];
    const float* W_out   = (const float*)d_in[6];
    float* out           = (float*)d_out;

    char* w = (char*)d_ws;
    size_t off = 0;
    auto alloc = [&](size_t bytes) -> void* {
        off = (off + 255) & ~(size_t)255;
        void* p = w + off;
        off += bytes;
        return p;
    };

    uint16_t* xb    = (uint16_t*)alloc((size_t)BT * D_MODEL * 2);
    uint16_t* w1b   = (uint16_t*)alloc((size_t)D_PROJ * D_MODEL * 2);
    uint16_t* w2b   = (uint16_t*)alloc((size_t)D_MODEL * D_INNER * 2);
    float*    proj  = (float*)   alloc((size_t)BT * D_PROJ * 4);
    float*    Bn    = (float*)   alloc((size_t)BT * 64 * 4);
    float*    Cn    = (float*)   alloc((size_t)BT * 64 * 4);
    float*    alpha = (float*)   alloc((size_t)BT * 32 * 4);
    float*    dtv   = (float*)   alloc((size_t)BT * 32 * 4);
    float*    lam   = (float*)   alloc((size_t)BT * 32 * 4);
    float*    dtm   = (float*)   alloc((size_t)BT * 4);
    float*    cosA  = (float*)   alloc((size_t)BT * 32 * 4);
    float*    sinA  = (float*)   alloc((size_t)BT * 32 * 4);
    float*    ybuf  = (float*)   alloc((size_t)BT * D_INNER * 4);
    uint16_t* yb    = (uint16_t*)alloc((size_t)BT * D_INNER * 2);

    // K0: conversions
    f32_to_bf16_kernel<<<2048, 256, 0, stream>>>(x,     xb,  (size_t)BT * D_MODEL);
    f32_to_bf16_kernel<<<2048, 256, 0, stream>>>(W_in,  w1b, (size_t)D_PROJ * D_MODEL);
    f32_to_bf16_kernel<<<1024, 256, 0, stream>>>(W_out, w2b, (size_t)D_MODEL * D_INNER);

    // K1: proj = x @ W_in^T   (M=8192, N=4320, K=1024)
    gemm_bf16_kernel<<<dim3((D_PROJ + BN - 1) / BN, BT / BM), 256, 0, stream>>>(
        xb, w1b, proj, BT, D_PROJ, D_MODEL);

    // K2: per-token prep, SiLU, serial angle cumsum, RoPE
    prep_pos_kernel<<<BT, 64, 0, stream>>>(proj, A_log, dt_bias, Bbias, Cbias,
                                           Bn, Cn, alpha, dtv, lam, dtm);
    silu_x_kernel<<<4096, 256, 0, stream>>>(proj);
    cumsum_theta_kernel<<<BATCH, 32, 0, stream>>>(proj, dtm, cosA, sinA);
    rope_kernel<<<1024, 256, 0, stream>>>(cosA, sinA, Bn, Cn);

    // K3: sequential scan, one block per (batch, head)
    scan_kernel<<<BATCH * N_HEADS, 256, 0, stream>>>(proj, Bn, Cn, alpha, dtv, lam, ybuf);

    // K4: gate + RMS-norm -> bf16
    gate_norm_kernel<<<BT, 256, 0, stream>>>(ybuf, proj, yb);

    // K5: out = y @ W_out^T   (M=8192, N=1024, K=2048)
    gemm_bf16_kernel<<<dim3(D_MODEL / BN, BT / BM), 256, 0, stream>>>(
        yb, w2b, out, BT, D_MODEL, D_INNER);
}
